// EndToEndRPModel_25658134626518
// MI455X (gfx1250) — matrix operand order repaired and checked by reading, not compiled, not run on hardware
//
#include <hip/hip_runtime.h>
#include <cstdint>

// ---------------------------------------------------------------------------
// CDNA5 / gfx1250 end-to-end RP model.
// B=64, F=8, T=512, D=6, RP=64.
// ---------------------------------------------------------------------------

typedef __attribute__((ext_vector_type(2)))  float    v2f;
typedef __attribute__((ext_vector_type(8)))  float    v8f;
typedef __attribute__((ext_vector_type(16))) _Float16 v16h;

#define BN_RSQRT 0.99999500003749972f   // 1/sqrt(1+1e-5)

__device__ __forceinline__ float gelu_(float x) {
    return 0.5f * x * (1.0f + erff(x * 0.70710678118654752f));
}

// ---------------------------------------------------------------------------
// K1: conv1d (8->32, k=7, pad=3) + BN + GELU.  out h1 (64,32,512)
// ---------------------------------------------------------------------------
__global__ void conv1d_1(const float* __restrict__ x, const float* __restrict__ w,
                         const float* __restrict__ g, const float* __restrict__ bb,
                         float* __restrict__ out) {
    int idx = blockIdx.x * blockDim.x + threadIdx.x;          // 64*32*512
    int t = idx & 511, oc = (idx >> 9) & 31, b = idx >> 14;
    const float* xb = x + (size_t)b * 8 * 512;
    const float* wr = w + (size_t)oc * 8 * 7;
    float acc = 0.f;
    #pragma unroll
    for (int ic = 0; ic < 8; ++ic)
        #pragma unroll
        for (int k = 0; k < 7; ++k) {
            int tt = t + k - 3;
            if (tt >= 0 && tt < 512) acc += xb[ic * 512 + tt] * wr[ic * 7 + k];
        }
    out[idx] = gelu_(acc * (g[oc] * BN_RSQRT) + bb[oc]);
}

// ---------------------------------------------------------------------------
// K2: conv1d (32->64, k=5, pad=2) + BN + GELU.  out h2 (64,64,512)
// ---------------------------------------------------------------------------
__global__ void conv1d_2(const float* __restrict__ in, const float* __restrict__ w,
                         const float* __restrict__ g, const float* __restrict__ bb,
                         float* __restrict__ out) {
    int idx = blockIdx.x * blockDim.x + threadIdx.x;          // 64*64*512
    int t = idx & 511, oc = (idx >> 9) & 63, b = idx >> 15;
    const float* hb = in + (size_t)b * 32 * 512;
    const float* wr = w + (size_t)oc * 32 * 5;
    float acc = 0.f;
    for (int ic = 0; ic < 32; ++ic)
        #pragma unroll
        for (int k = 0; k < 5; ++k) {
            int tt = t + k - 2;
            if (tt >= 0 && tt < 512) acc += hb[ic * 512 + tt] * wr[ic * 5 + k];
        }
    out[idx] = gelu_(acc * (g[oc] * BN_RSQRT) + bb[oc]);
}

// ---------------------------------------------------------------------------
// K3: conv1d (64->6, k=3, pad=1) -> zt (64,512,8) D padded to 8 with zeros,
//     plus sq[b][t] = sum_d z^2.
// ---------------------------------------------------------------------------
__global__ void conv1d_3(const float* __restrict__ in, const float* __restrict__ w,
                         float* __restrict__ zt, float* __restrict__ sq) {
    int idx = blockIdx.x * blockDim.x + threadIdx.x;          // 64*512
    int t = idx & 511, b = idx >> 9;
    const float* hb = in + (size_t)b * 64 * 512;
    float z[6];
    #pragma unroll
    for (int d = 0; d < 6; ++d) z[d] = 0.f;
    for (int ic = 0; ic < 64; ++ic) {
        float x0 = (t - 1 >= 0) ? hb[ic * 512 + t - 1] : 0.f;
        float x1 = hb[ic * 512 + t];
        float x2 = (t + 1 < 512) ? hb[ic * 512 + t + 1] : 0.f;
        #pragma unroll
        for (int d = 0; d < 6; ++d) {
            const float* wr = w + (size_t)d * 64 * 3 + ic * 3;
            z[d] += x0 * wr[0] + x1 * wr[1] + x2 * wr[2];
        }
    }
    float s = 0.f;
    float* zrow = zt + (size_t)idx * 8;
    #pragma unroll
    for (int d = 0; d < 6; ++d) { zrow[d] = z[d]; s += z[d] * z[d]; }
    zrow[6] = 0.f; zrow[7] = 0.f;
    sq[idx] = s;
}

// ---------------------------------------------------------------------------
// K-init: min/max accumulators (positive floats -> uint ordering works).
// ---------------------------------------------------------------------------
__global__ void init_stats(unsigned* mn, unsigned* mx) {
    int i = threadIdx.x;                                      // 64
    mn[i] = 0x7f800000u; mx[i] = 0u;
}

// ---------------------------------------------------------------------------
// K4: gram via fp32 WMMA (16x16x4, K=8 in 2 steps), fused dist + per-tile sum.
// One wave per 16x16 tile; 64 batches * 32*32 tiles = 65536 waves.
// partial[b*1024 + tile] = sum of 256 dist values in tile.
// ---------------------------------------------------------------------------
__global__ void gram_sigma(const float* __restrict__ zt, const float* __restrict__ sq,
                           float* __restrict__ partial) {
    int lane = threadIdx.x & 31;
    int wave = blockIdx.x * (blockDim.x >> 5) + (threadIdx.x >> 5);
    int b    = wave >> 10;
    int tile = wave & 1023;
    int ti = tile >> 5, tj = tile & 31;
    int l15 = lane & 15;
    int hi  = lane >> 4;                      // 0: K={0,1}/{4,5}, 1: K={2,3}/{6,7}
    int m = ti * 16 + l15;
    int n = tj * 16 + l15;
    const float* za = zt + ((size_t)b * 512 + m) * 8 + hi * 2;
    const float* zb = zt + ((size_t)b * 512 + n) * 8 + hi * 2;
    v2f a0 = *(const v2f*)(za);
    v2f b0 = *(const v2f*)(zb);
    v2f a1 = *(const v2f*)(za + 4);
    v2f b1 = *(const v2f*)(zb + 4);
    v8f c = {};
    c = __builtin_amdgcn_wmma_f32_16x16x4_f32(false, a0, false, b0, (short)0, c, false, false);
    c = __builtin_amdgcn_wmma_f32_16x16x4_f32(false, a1, false, b1, (short)0, c, false, false);
    const float* sqb = sq + (size_t)b * 512;
    float sn = sqb[n];
    int mrow = ti * 16 + hi * 8;              // C layout: lanes>=16 hold M = r+8
    float lsum = 0.f;
    #pragma unroll
    for (int r = 0; r < 8; ++r) {
        float d2 = fmaxf(sqb[mrow + r] + sn - 2.0f * c[r], 0.0f);
        lsum += sqrtf(d2 + 1e-6f);
    }
    #pragma unroll
    for (int off = 16; off > 0; off >>= 1) lsum += __shfl_xor(lsum, off, 32);
    if (lane == 0) partial[(size_t)b * 1024 + tile] = lsum;
}

// ---------------------------------------------------------------------------
// K5: deterministic sigma reduction.  sigma[b] = mean(dist) + 1e-4
// ---------------------------------------------------------------------------
__global__ void sigma_reduce(const float* __restrict__ partial, float* __restrict__ sigma) {
    __shared__ float red[256];
    int b = blockIdx.x, t = threadIdx.x;
    float s = 0.f;
    for (int i = t; i < 1024; i += 256) s += partial[(size_t)b * 1024 + i];
    red[t] = s; __syncthreads();
    for (int k = 128; k > 0; k >>= 1) { if (t < k) red[t] += red[t + k]; __syncthreads(); }
    if (t == 0) sigma[b] = red[0] * (1.0f / 262144.0f) + 1e-4f;
}

// ---------------------------------------------------------------------------
// K6: resized RP map.  Bilinear 512->64 has constant coords: rows {8r+3,8r+4},
// weight 0.5 each -> average of 4 rp values.  Recompute dist at just those
// points (K=6 dot) instead of materializing 512x512.
// ---------------------------------------------------------------------------
__global__ void rp_resize(const float* __restrict__ zt, const float* __restrict__ sq,
                          const float* __restrict__ sigma, float* __restrict__ rp64,
                          unsigned* __restrict__ mnb, unsigned* __restrict__ mxb) {
    int idx = blockIdx.x * blockDim.x + threadIdx.x;          // 64*64*64
    int cc = idx & 63, rr = (idx >> 6) & 63, b = idx >> 12;
    float sg = sigma[b];
    const float* Z = zt + (size_t)b * 512 * 8;
    const float* S = sq + (size_t)b * 512;
    int is[2] = { 8 * rr + 3, 8 * rr + 4 };
    int js[2] = { 8 * cc + 3, 8 * cc + 4 };
    float acc = 0.f;
    #pragma unroll
    for (int ai = 0; ai < 2; ++ai)
        #pragma unroll
        for (int ci = 0; ci < 2; ++ci) {
            const float* zi = Z + is[ai] * 8;
            const float* zj = Z + js[ci] * 8;
            float gdot = 0.f;
            #pragma unroll
            for (int d = 0; d < 6; ++d) gdot += zi[d] * zj[d];
            float d2 = fmaxf(S[is[ai]] + S[js[ci]] - 2.f * gdot, 0.f);
            float dist = sqrtf(d2 + 1e-6f);
            acc += expf(fmaxf(-dist / sg, -20.0f));
        }
    float v = 0.25f * acc;                                    // positive
    rp64[idx] = v;
    atomicMin(&mnb[b], __float_as_uint(v));
    atomicMax(&mxb[b], __float_as_uint(v));
}

// ---------------------------------------------------------------------------
// K7: min/max normalize.
// ---------------------------------------------------------------------------
__global__ void rp_norm_k(const float* __restrict__ rp64, const unsigned* __restrict__ mnb,
                          const unsigned* __restrict__ mxb, float* __restrict__ rpn) {
    int idx = blockIdx.x * blockDim.x + threadIdx.x;          // 64*64*64
    int b = idx >> 12;
    float mn = __uint_as_float(mnb[b]);
    float mx = __uint_as_float(mxb[b]);
    rpn[idx] = (rp64[idx] - mn) / (mx - mn + 1e-4f);
}

// ---------------------------------------------------------------------------
// Fused conv3x3(pad1)+BN+GELU+maxpool2.  One thread = one pooled output:
// loads a 4x4 patch per input channel, produces 4 conv outputs, pools.
// ---------------------------------------------------------------------------
template <int IC, int OC, int S>
__global__ void conv3x3_bn_gelu_pool(const float* __restrict__ in, const float* __restrict__ w,
                                     const float* __restrict__ g, const float* __restrict__ bb,
                                     float* __restrict__ out) {
    constexpr int O = S / 2;
    int idx = blockIdx.x * blockDim.x + threadIdx.x;          // 64*OC*O*O
    int x = idx % O, y = (idx / O) % O, oc = (idx / (O * O)) % OC, b = idx / (O * O * OC);
    int y0 = 2 * y, x0 = 2 * x;
    float a00 = 0.f, a01 = 0.f, a10 = 0.f, a11 = 0.f;
    for (int ic = 0; ic < IC; ++ic) {
        const float* p = in + ((size_t)b * IC + ic) * S * S;
        float patch[4][4];
        #pragma unroll
        for (int r = 0; r < 4; ++r) {
            int iy = y0 - 1 + r;
            #pragma unroll
            for (int cidx = 0; cidx < 4; ++cidx) {
                int ix = x0 - 1 + cidx;
                patch[r][cidx] = (iy >= 0 && iy < S && ix >= 0 && ix < S) ? p[iy * S + ix] : 0.f;
            }
        }
        const float* wr = w + ((size_t)oc * IC + ic) * 9;
        float w9[9];
        #pragma unroll
        for (int i = 0; i < 9; ++i) w9[i] = wr[i];
        #pragma unroll
        for (int ky = 0; ky < 3; ++ky)
            #pragma unroll
            for (int kx = 0; kx < 3; ++kx) {
                float wv = w9[ky * 3 + kx];
                a00 += patch[ky][kx] * wv;
                a01 += patch[ky][kx + 1] * wv;
                a10 += patch[ky + 1][kx] * wv;
                a11 += patch[ky + 1][kx + 1] * wv;
            }
    }
    float sc = g[oc] * BN_RSQRT, bi = bb[oc];
    out[idx] = fmaxf(fmaxf(gelu_(a00 * sc + bi), gelu_(a01 * sc + bi)),
                     fmaxf(gelu_(a10 * sc + bi), gelu_(a11 * sc + bi)));
}

// ---------------------------------------------------------------------------
// K11: conv c4 (128->128, 8x8) + BN + GELU + avgpool2 -> f16 features (64,2048)
// flatten order: c*16 + y*4 + x.
// ---------------------------------------------------------------------------
__global__ void conv4_avg_feat(const float* __restrict__ in, const float* __restrict__ w,
                               const float* __restrict__ g, const float* __restrict__ bb,
                               _Float16* __restrict__ a16) {
    int idx = blockIdx.x * blockDim.x + threadIdx.x;          // 64*128*16
    int x = idx & 3, y = (idx >> 2) & 3, oc = (idx >> 4) & 127, b = idx >> 11;
    int y0 = 2 * y, x0 = 2 * x;
    float a00 = 0.f, a01 = 0.f, a10 = 0.f, a11 = 0.f;
    for (int ic = 0; ic < 128; ++ic) {
        const float* p = in + ((size_t)b * 128 + ic) * 64;
        float patch[4][4];
        #pragma unroll
        for (int r = 0; r < 4; ++r) {
            int iy = y0 - 1 + r;
            #pragma unroll
            for (int cidx = 0; cidx < 4; ++cidx) {
                int ix = x0 - 1 + cidx;
                patch[r][cidx] = (iy >= 0 && iy < 8 && ix >= 0 && ix < 8) ? p[iy * 8 + ix] : 0.f;
            }
        }
        const float* wr = w + ((size_t)oc * 128 + ic) * 9;
        float w9[9];
        #pragma unroll
        for (int i = 0; i < 9; ++i) w9[i] = wr[i];
        #pragma unroll
        for (int ky = 0; ky < 3; ++ky)
            #pragma unroll
            for (int kx = 0; kx < 3; ++kx) {
                float wv = w9[ky * 3 + kx];
                a00 += patch[ky][kx] * wv;
                a01 += patch[ky][kx + 1] * wv;
                a10 += patch[ky + 1][kx] * wv;
                a11 += patch[ky + 1][kx + 1] * wv;
            }
    }
    float sc = g[oc] * BN_RSQRT, bi = bb[oc];
    float v = 0.25f * (gelu_(a00 * sc + bi) + gelu_(a01 * sc + bi) +
                       gelu_(a10 * sc + bi) + gelu_(a11 * sc + bi));
    a16[(size_t)b * 2048 + oc * 16 + y * 4 + x] = (_Float16)v;
}

// ---------------------------------------------------------------------------
// K12a: fc1 weights -> f16 (row-major [n][k] matches B-fragment gather).
// ---------------------------------------------------------------------------
__global__ void wcvt(const float* __restrict__ w, _Float16* __restrict__ w16) {
    int idx = blockIdx.x * blockDim.x + threadIdx.x;          // 256*2048
    w16[idx] = (_Float16)w[idx];
}

// ---------------------------------------------------------------------------
// K12: fc1 GEMM via v_wmma_f32_16x16x32_f16.  M=64 (batch), N=256, K=2048.
// One wave per 16x16 tile, 64 waves total.  Fragment gathers follow the
// documented 16-bit A (16x32) and B (32x16) wave32 VGPR layouts.
// ---------------------------------------------------------------------------
__global__ void fc1_wmma(const _Float16* __restrict__ a16, const _Float16* __restrict__ w16,
                         const float* __restrict__ fc1_b, float* __restrict__ hfc) {
    int lane = threadIdx.x & 31;
    int wave = blockIdx.x * (blockDim.x >> 5) + (threadIdx.x >> 5);  // 0..63
    int mi = wave >> 4, ni = wave & 15;
    int l15 = lane & 15;
    int baseA = (lane >> 4) ? 8 : 0;          // A: lanes>=16 hold K+8
    int baseB = (lane >> 4) ? 16 : 0;         // unused: B gathers use baseA (K+8)
    const _Float16* arow = a16 + (size_t)(mi * 16 + l15) * 2048;
    const _Float16* brow = w16 + (size_t)(ni * 16 + l15) * 2048;
    v8f c = {};
    for (int k0 = 0; k0 < 2048; k0 += 32) {
        __builtin_prefetch(arow + k0 + 64, 0, 1);
        __builtin_prefetch(brow + k0 + 64, 0, 1);
        v16h a, bf;
        #pragma unroll
        for (int p = 0; p < 8; ++p) {
            int ka = k0 + 2 * p + (p < 4 ? 0 : 8) + baseA;    // VGPRs 0-3: K 0..7, 4-7: K 16..23
            union { float f; _Float16 h[2]; } ua;
            ua.f = *(const float*)(arow + ka);
            a[2 * p] = ua.h[0]; a[2 * p + 1] = ua.h[1];
            int kb = k0 + 2 * p + (p < 4 ? 0 : 8) + baseA;    // same K order as A
            union { float f; _Float16 h[2]; } ub;
            ub.f = *(const float*)(brow + kb);
            bf[2 * p] = ub.h[0]; bf[2 * p + 1] = ub.h[1];
        }
        c = __builtin_amdgcn_wmma_f32_16x16x32_f16(false, a, false, bf, (short)0, c, false, false);
    }
    #pragma unroll
    for (int r = 0; r < 8; ++r) {
        int mm = mi * 16 + r + baseA;                         // C: lanes>=16 hold M=r+8
        int nn = ni * 16 + l15;
        hfc[(size_t)mm * 256 + nn] = gelu_(c[r] + fc1_b[nn]);
    }
}

// ---------------------------------------------------------------------------
// K13: fc2 (256->1) per-batch reduction.  Inputs staged through LDS with
// CDNA5 async global->LDS copies (ASYNCcnt-tracked, s_wait_asynccnt).
// Layout: sh[0:256) = hfc row b, sh[256:512) = fc2_w.  One B128 per lane,
// four issuing waves per block; all addresses 16B-aligned.
// ---------------------------------------------------------------------------
typedef __attribute__((address_space(3))) float lds_f32;

__global__ void fc2_out(const float* __restrict__ hfc, const float* __restrict__ fc2_w,
                        const float* __restrict__ fc2_b, float* __restrict__ out) {
    __shared__ __align__(16) float sh[512];
    int b = blockIdx.x, t = threadIdx.x;
    unsigned lds_base = (unsigned)(size_t)(lds_f32*)&sh[0];
    if (t < 128) {
        int slot = t & 63;                                    // 16B chunk index
        const float* src = (t < 64) ? (hfc + (size_t)b * 256 + slot * 4)
                                    : (fc2_w + slot * 4);
        unsigned lds_off = lds_base + ((t < 64) ? 0u : 1024u) + (unsigned)slot * 16u;
        asm volatile("global_load_async_to_lds_b128 %0, %1, off"
                     :: "v"(lds_off), "v"(src) : "memory");
    }
    asm volatile("s_wait_asynccnt 0" ::: "memory");
    __syncthreads();
    float v = sh[t] * sh[256 + t];
    sh[t] = v;                                                // safe: each thread owns sh[t]
    __syncthreads();
    for (int k = 128; k > 0; k >>= 1) { if (t < k) sh[t] += sh[t + k]; __syncthreads(); }
    if (t == 0) out[b] = sh[0] + fc2_b[0];
}

// ---------------------------------------------------------------------------
// Workspace layout (bytes, 256-aligned blocks)
// ---------------------------------------------------------------------------
static constexpr size_t OFF_H1   = 0;                          // 64*32*512 f   (4 MB)
static constexpr size_t OFF_H2   = OFF_H1  + (size_t)4  * 1024 * 1024;  // 8 MB
static constexpr size_t OFF_ZT   = OFF_H2  + (size_t)8  * 1024 * 1024;  // 1 MB
static constexpr size_t OFF_SQ   = OFF_ZT  + (size_t)1  * 1024 * 1024;  // 128 KB
static constexpr size_t OFF_PART = OFF_SQ  + 128 * 1024;                // 256 KB
static constexpr size_t OFF_SIG  = OFF_PART + 256 * 1024;               // 256 B
static constexpr size_t OFF_MN   = OFF_SIG + 256;
static constexpr size_t OFF_MX   = OFF_MN  + 256;
static constexpr size_t OFF_RP   = OFF_MX  + 256;                       // 1 MB
static constexpr size_t OFF_RPN  = OFF_RP  + (size_t)1 * 1024 * 1024;   // 1 MB
static constexpr size_t OFF_P1   = OFF_RPN + (size_t)1 * 1024 * 1024;   // 8 MB
static constexpr size_t OFF_P2   = OFF_P1  + (size_t)8 * 1024 * 1024;   // 4 MB
static constexpr size_t OFF_P3   = OFF_P2  + (size_t)4 * 1024 * 1024;   // 2 MB
static constexpr size_t OFF_A16  = OFF_P3  + (size_t)2 * 1024 * 1024;   // 256 KB
static constexpr size_t OFF_W16  = OFF_A16 + 256 * 1024;                // 1 MB
static constexpr size_t OFF_HFC  = OFF_W16 + (size_t)1 * 1024 * 1024;   // 64 KB

extern "C" void kernel_launch(void* const* d_in, const int* in_sizes, int n_in,
                              void* d_out, int out_size, void* d_ws, size_t ws_size,
                              hipStream_t stream) {
    const float* x     = (const float*)d_in[0];
    const float* w1    = (const float*)d_in[1];
    const float* g1    = (const float*)d_in[2];
    const float* b1    = (const float*)d_in[3];
    const float* w2    = (const float*)d_in[4];
    const float* g2    = (const float*)d_in[5];
    const float* b2    = (const float*)d_in[6];
    const float* w3    = (const float*)d_in[7];
    const float* c1    = (const float*)d_in[8];
    const float* cg1   = (const float*)d_in[9];
    const float* cb1   = (const float*)d_in[10];
    const float* c2    = (const float*)d_in[11];
    const float* cg2   = (const float*)d_in[12];
    const float* cb2   = (const float*)d_in[13];
    const float* c3    = (const float*)d_in[14];
    const float* cg3   = (const float*)d_in[15];
    const float* cb3   = (const float*)d_in[16];
    const float* c4    = (const float*)d_in[17];
    const float* cg4   = (const float*)d_in[18];
    const float* cb4   = (const float*)d_in[19];
    const float* fc1_w = (const float*)d_in[20];
    const float* fc1_b = (const float*)d_in[21];
    const float* fc2_w = (const float*)d_in[22];
    const float* fc2_b = (const float*)d_in[23];
    float* out = (float*)d_out;

    char* ws = (char*)d_ws;
    float*     h1    = (float*)(ws + OFF_H1);
    float*     h2    = (float*)(ws + OFF_H2);
    float*     zt    = (float*)(ws + OFF_ZT);
    float*     sq    = (float*)(ws + OFF_SQ);
    float*     part  = (float*)(ws + OFF_PART);
    float*     sigma = (float*)(ws + OFF_SIG);
    unsigned*  mnb   = (unsigned*)(ws + OFF_MN);
    unsigned*  mxb   = (unsigned*)(ws + OFF_MX);
    float*     rp64  = (float*)(ws + OFF_RP);
    float*     rpn   = (float*)(ws + OFF_RPN);
    float*     p1    = (float*)(ws + OFF_P1);
    float*     p2    = (float*)(ws + OFF_P2);
    float*     p3    = (float*)(ws + OFF_P3);
    _Float16*  a16   = (_Float16*)(ws + OFF_A16);
    _Float16*  w16   = (_Float16*)(ws + OFF_W16);
    float*     hfc   = (float*)(ws + OFF_HFC);

    // 1D conv stack
    conv1d_1<<<4096, 256, 0, stream>>>(x, w1, g1, b1, h1);
    conv1d_2<<<8192, 256, 0, stream>>>(h1, w2, g2, b2, h2);
    conv1d_3<<<128, 256, 0, stream>>>(h2, w3, zt, sq);

    // RP map: WMMA gram + sigma, constant-coefficient resize, normalize
    init_stats<<<1, 64, 0, stream>>>(mnb, mxb);
    gram_sigma<<<8192, 256, 0, stream>>>(zt, sq, part);        // fp32 WMMA
    sigma_reduce<<<64, 256, 0, stream>>>(part, sigma);
    rp_resize<<<1024, 256, 0, stream>>>(zt, sq, sigma, rp64, mnb, mxb);
    rp_norm_k<<<1024, 256, 0, stream>>>(rp64, mnb, mxb, rpn);

    // 2D conv stack (fused conv+BN+GELU+pool)
    conv3x3_bn_gelu_pool<1, 32, 64><<<8192, 256, 0, stream>>>(rpn, c1, cg1, cb1, p1);
    conv3x3_bn_gelu_pool<32, 64, 32><<<4096, 256, 0, stream>>>(p1, c2, cg2, cb2, p2);
    conv3x3_bn_gelu_pool<64, 128, 16><<<2048, 256, 0, stream>>>(p2, c3, cg3, cb3, p3);
    conv4_avg_feat<<<512, 256, 0, stream>>>(p3, c4, cg4, cb4, a16);

    // Head: f16 WMMA GEMM + async-LDS-staged reduction
    wcvt<<<2048, 256, 0, stream>>>(fc1_w, w16);
    fc1_wmma<<<16, 128, 0, stream>>>(a16, w16, fc1_b, hfc);    // f16 WMMA
    fc2_out<<<64, 256, 0, stream>>>(hfc, fc2_w, fc2_b, out);
}
